// DHMNN_84189948936332
// MI455X (gfx1250) — matrix operand order repaired and checked by reading, not compiled, not run on hardware
//
#include <hip/hip_runtime.h>
#include <math.h>

// ---------------- problem constants (from reference) ----------------
#define NGc     4096
#define NVc     50000
#define NEc     50000
#define EDGc    300000
#define NTc     300000
#define NHc     300000
#define Hc      256
#define DGc     256
#define DLc     128
#define HEADSc  4
#define DHc     64
#define NGRAPHc 64
#define HIDc    256

typedef __attribute__((ext_vector_type(16))) _Float16 v16h;
typedef __attribute__((ext_vector_type(8)))  float    v8f;

union U16H { v16h v; _Float16 h[16]; };

// ---------------- block reductions (256 threads, pow2) ----------------
__device__ __forceinline__ float block_sum(float v, float* sm) {
    int t = threadIdx.x;
    sm[t] = v; __syncthreads();
    for (int s = blockDim.x >> 1; s > 0; s >>= 1) {
        if (t < s) sm[t] += sm[t + s];
        __syncthreads();
    }
    float r = sm[0]; __syncthreads();
    return r;
}
__device__ __forceinline__ float block_maxr(float v, float* sm) {
    int t = threadIdx.x;
    sm[t] = v; __syncthreads();
    for (int s = blockDim.x >> 1; s > 0; s >>= 1) {
        if (t < s) sm[t] = fmaxf(sm[t], sm[t + s]);
        __syncthreads();
    }
    float r = sm[0]; __syncthreads();
    return r;
}

__device__ __forceinline__ void atomicMaxF(float* addr, float val) {
    if (val >= 0.f) atomicMax((int*)addr, __float_as_int(val));
    else            atomicMin((unsigned int*)addr, __float_as_uint(val));
}

// ---------------- generic WMMA GEMM: Y = act(scale*(A @ B^T) + bias) --------
// A: MxK (row stride lda). B: NxK (row stride ldb) if BT==0,
//    else B element (n,k) = B[k*ldb + n] (K-major, for attention*V).
// One wave computes a 16x64 strip (4 wmma accumulators sharing one A frag);
// 4 waves (128 thr) per block -> block covers 64 rows x 64 cols.
// LDB_CT != 0 makes the B stride a compile-time constant so the BT path
// uses immediate-offset global loads instead of 64-bit address chains.
template <int BT, int LDB_CT>
__global__ void k_gemm_t(const float* __restrict__ A, int lda,
                         const float* __restrict__ B, int ldb_rt,
                         const float* __restrict__ bias,
                         float* __restrict__ Y, int ldy,
                         int M, int N, int K, float scale, int act) {
    const int ldb = LDB_CT ? LDB_CT : ldb_rt;
    int wave = threadIdx.x >> 5;
    int lane = threadIdx.x & 31;
    int nb0 = blockIdx.x << 6;                 // 64 columns per wave
    int mb0 = (blockIdx.y * 4 + wave) << 4;    // 16 rows per wave
    if (mb0 >= M) return;
    int lh = lane & 15, hi = lane >> 4;

    const float* Arow = A + (size_t)(mb0 + lh) * lda;
    v8f c[4];
    #pragma unroll
    for (int t = 0; t < 4; ++t) c[t] = (v8f){0.f,0.f,0.f,0.f,0.f,0.f,0.f,0.f};

    for (int kb = 0; kb < K; kb += 32) {
        // A fragment: element i -> K = kb + 16*hi + i
        U16H af;
        const float* ap = Arow + kb + hi * 16;
        float4 a0 = *(const float4*)(ap + 0);
        float4 a1 = *(const float4*)(ap + 4);
        float4 a2 = *(const float4*)(ap + 8);
        float4 a3 = *(const float4*)(ap + 12);
        af.h[0]=(_Float16)a0.x; af.h[1]=(_Float16)a0.y; af.h[2]=(_Float16)a0.z; af.h[3]=(_Float16)a0.w;
        af.h[4]=(_Float16)a1.x; af.h[5]=(_Float16)a1.y; af.h[6]=(_Float16)a1.z; af.h[7]=(_Float16)a1.w;
        af.h[8]=(_Float16)a2.x; af.h[9]=(_Float16)a2.y; af.h[10]=(_Float16)a2.z; af.h[11]=(_Float16)a2.w;
        af.h[12]=(_Float16)a3.x; af.h[13]=(_Float16)a3.y; af.h[14]=(_Float16)a3.z; af.h[15]=(_Float16)a3.w;

        #pragma unroll
        for (int t = 0; t < 4; ++t) {
            // B fragment: element i -> K = kb + 16*hi + i, N = nb + lh
            U16H bf;
            int nb = nb0 + t * 16;
            if (!BT) {
                const float* bp = B + (size_t)(nb + lh) * ldb + kb + hi * 16;
                float4 b0 = *(const float4*)(bp + 0);
                float4 b1 = *(const float4*)(bp + 4);
                float4 b2 = *(const float4*)(bp + 8);
                float4 b3 = *(const float4*)(bp + 12);
                bf.h[0]=(_Float16)b0.x; bf.h[1]=(_Float16)b0.y; bf.h[2]=(_Float16)b0.z; bf.h[3]=(_Float16)b0.w;
                bf.h[4]=(_Float16)b1.x; bf.h[5]=(_Float16)b1.y; bf.h[6]=(_Float16)b1.z; bf.h[7]=(_Float16)b1.w;
                bf.h[8]=(_Float16)b2.x; bf.h[9]=(_Float16)b2.y; bf.h[10]=(_Float16)b2.z; bf.h[11]=(_Float16)b2.w;
                bf.h[12]=(_Float16)b3.x; bf.h[13]=(_Float16)b3.y; bf.h[14]=(_Float16)b3.z; bf.h[15]=(_Float16)b3.w;
            } else {
                const float* bp = B + (size_t)(kb + hi * 16) * ldb + nb + lh;
                #pragma unroll
                for (int i = 0; i < 16; ++i) bf.h[i] = (_Float16)bp[(size_t)i * ldb];
            }
            c[t] = __builtin_amdgcn_wmma_f32_16x16x32_f16(false, af.v, false, bf.v,
                                                          (short)0, c[t], false, false);
        }
    }
    #pragma unroll
    for (int t = 0; t < 4; ++t) {
        int col = nb0 + t * 16 + lh;
        float bv = bias ? bias[col] : 0.f;
        #pragma unroll
        for (int r = 0; r < 8; ++r) {
            float v = c[t][r] * scale + bv;
            if (act == 1)      v = tanhf(v);
            else if (act == 2) v = fmaxf(v, 0.f);
            Y[(size_t)(mb0 + r + hi * 8) * ldy + col] = v;
        }
    }
}

// ---------------- elementwise / rowwise kernels ----------------
__global__ void k_fill(float* __restrict__ p, float v, long n) {
    long i = (long)blockIdx.x * blockDim.x + threadIdx.x;
    if (i < n) p[i] = v;
}

__global__ void k_softmax(float* __restrict__ S, int n) {
    __shared__ float sm[256];
    float* r = S + (size_t)blockIdx.x * n;
    float m = -3.402823e38f;
    for (int j = threadIdx.x; j < n; j += blockDim.x) m = fmaxf(m, r[j]);
    m = block_maxr(m, sm);
    float s = 0.f;
    for (int j = threadIdx.x; j < n; j += blockDim.x) {
        float e = __expf(r[j] - m); r[j] = e; s += e;
    }
    s = block_sum(s, sm);
    float inv = 1.f / s;
    for (int j = threadIdx.x; j < n; j += blockDim.x) r[j] *= inv;
}

// Y = tanh(LN(X + R)) , one 256-thread block per row (H==blockDim)
__global__ void k_add_ln_tanh(const float* __restrict__ X, const float* __restrict__ R,
                              const float* __restrict__ w, const float* __restrict__ b,
                              float* __restrict__ Y, int H) {
    __shared__ float sm[256];
    size_t base = (size_t)blockIdx.x * H + threadIdx.x;
    float v = X[base] + R[base];
    float mu = block_sum(v, sm) / (float)H;
    float d = v - mu;
    float var = block_sum(d * d, sm) / (float)H;
    Y[base] = tanhf(d * rsqrtf(var + 1e-5f) * w[threadIdx.x] + b[threadIdx.x]);
}

// in-place LN per row
__global__ void k_ln_rows(float* __restrict__ X, const float* __restrict__ w,
                          const float* __restrict__ b, int H) {
    __shared__ float sm[256];
    size_t base = (size_t)blockIdx.x * H + threadIdx.x;
    float v = X[base];
    float mu = block_sum(v, sm) / (float)H;
    float d = v - mu;
    float var = block_sum(d * d, sm) / (float)H;
    X[base] = d * rsqrtf(var + 1e-5f) * w[threadIdx.x] + b[threadIdx.x];
}

// ---------------- segment mean machinery ----------------
// one block per index t; gathers X[cmap?cmap[vert[t]]:vert[t]], scatters to Msum[edge[t]]
__global__ void k_edge_mean_acc(const float* __restrict__ X, const int* __restrict__ cmap,
                                const int* __restrict__ vert, const int* __restrict__ edge,
                                float* __restrict__ Msum, float* __restrict__ cnt, int H) {
    int t = blockIdx.x, j = threadIdx.x;
    int v = vert[t];
    if (cmap) v = cmap[v];
    int e = edge[t];
    atomicAdd(&Msum[(size_t)e * H + j], X[(size_t)v * H + j]);
    if (j == 0) atomicAdd(&cnt[e], 1.f);
}

__global__ void k_div_rows(float* __restrict__ M, const float* __restrict__ cnt, int H) {
    int e = blockIdx.x, j = threadIdx.x;
    M[(size_t)e * H + j] /= fmaxf(cnt[e], 1.f);
}

// acc1[e] += sum_j (x-M1[e])^2 * SW[off1+j] ; acc2[e] += sum_j (x-M2[e])^2 * SW[off2+j]
__global__ void k_score_var(const float* __restrict__ X, const int* __restrict__ cmap,
                            const int* __restrict__ vert, const int* __restrict__ edge,
                            const float* __restrict__ M1, const float* __restrict__ M2,
                            const float* __restrict__ SW, int off1, int off2,
                            float* __restrict__ acc1, float* __restrict__ acc2, int H) {
    __shared__ float sm[256];
    int t = blockIdx.x, j = threadIdx.x;
    int v = vert[t];
    if (cmap) v = cmap[v];
    int e = edge[t];
    float x = X[(size_t)v * H + j];
    float d1 = x - M1[(size_t)e * H + j];
    float d2 = x - M2[(size_t)e * H + j];
    float s1 = block_sum(d1 * d1 * SW[off1 + j], sm);
    float s2 = block_sum(d2 * d2 * SW[off2 + j], sm);
    if (j == 0) { atomicAdd(&acc1[e], s1); atomicAdd(&acc2[e], s2); }
}

__global__ void k_se_final(const float* accT, const float* accHm, const float* accTH,
                           const float* accHT, const float* cntT, const float* cntH,
                           const float* sb, float* Se, int n) {
    int e = blockIdx.x * blockDim.x + threadIdx.x;
    if (e >= n) return;
    float ct = fmaxf(cntT[e], 1.f), ch = fmaxf(cntH[e], 1.f);
    float x = accT[e] / ct + accHm[e] / ch + accTH[e] / ct + accHT[e] / ch + sb[0];
    Se[e] = 1.f / (1.f + __expf(-x));
}

__global__ void k_cos_de(const float* __restrict__ A, const float* __restrict__ Mh,
                         float* __restrict__ De, int H) {
    __shared__ float sm[256];
    size_t base = (size_t)blockIdx.x * H + threadIdx.x;
    float a = A[base], b = Mh[base];
    float dot = block_sum(a * b, sm);
    float na = block_sum(a * a, sm);
    float nb = block_sum(b * b, sm);
    if (threadIdx.x == 0) {
        float c = dot / (fmaxf(sqrtf(na), 1e-8f) * fmaxf(sqrtf(nb), 1e-8f));
        De[blockIdx.x] = (c + 1.f) * 0.5f;
    }
}

// ---------------- graphnorm ----------------
__global__ void k_gn_acc(const float* __restrict__ X, const int* __restrict__ batch,
                         float* __restrict__ Msum, float* __restrict__ cnt, int H) {
    int v = blockIdx.x, j = threadIdx.x;
    int g = batch[v];
    atomicAdd(&Msum[(size_t)g * H + j], X[(size_t)v * H + j]);
    if (j == 0) atomicAdd(&cnt[g], 1.f);
}
__global__ void k_gn_var(const float* __restrict__ X, const int* __restrict__ batch,
                         const float* __restrict__ Mean, const float* __restrict__ ms,
                         float* __restrict__ Var, int H) {
    int v = blockIdx.x, j = threadIdx.x;
    int g = batch[v];
    float o = X[(size_t)v * H + j] - Mean[(size_t)g * H + j] * ms[j];
    atomicAdd(&Var[(size_t)g * H + j], o * o);
}
__global__ void k_gn_final(const float* __restrict__ X, const int* __restrict__ batch,
                           const float* __restrict__ Mean, const float* __restrict__ Var,
                           const float* __restrict__ ms, const float* __restrict__ w,
                           const float* __restrict__ b, float* __restrict__ Y, int H) {
    int v = blockIdx.x, j = threadIdx.x;
    int g = batch[v];
    float o = X[(size_t)v * H + j] - Mean[(size_t)g * H + j] * ms[j];
    Y[(size_t)v * H + j] = o * rsqrtf(Var[(size_t)g * H + j] + 1e-5f) * w[j] + b[j];
}

// ---------------- GAT ----------------
__global__ void k_gat_coef(const float* __restrict__ XW, const float* __restrict__ asrc,
                           const float* __restrict__ adst, float* __restrict__ AS,
                           float* __restrict__ AD, int H) {
    __shared__ float s1[256], s2[256];
    int v = blockIdx.x, j = threadIdx.x;
    float x = XW[(size_t)v * H + j];
    s1[j] = x * asrc[j];
    s2[j] = x * adst[j];
    __syncthreads();
    for (int s = 32; s > 0; s >>= 1) {
        if ((j & 63) < s) { s1[j] += s1[j + s]; s2[j] += s2[j + s]; }
        __syncthreads();
    }
    if ((j & 63) == 0) { AS[v * 4 + (j >> 6)] = s1[j]; AD[v * 4 + (j >> 6)] = s2[j]; }
}

__device__ __forceinline__ void gat_edge(const int* eidx, int i, int E, int& src, int& dst) {
    src = (i < E) ? eidx[i] : (i - E);
    dst = (i < E) ? eidx[E + i] : (i - E);
}
__device__ __forceinline__ float lrelu(float l) { return (l > 0.f) ? l : 0.2f * l; }

__global__ void k_gat_max(const int* __restrict__ eidx, const float* __restrict__ AS,
                          const float* __restrict__ AD, float* __restrict__ EMAX,
                          int E, int Nv) {
    int i = blockIdx.x * blockDim.x + threadIdx.x;
    if (i >= E + Nv) return;
    int src, dst; gat_edge(eidx, i, E, src, dst);
    #pragma unroll
    for (int h = 0; h < 4; ++h)
        atomicMaxF(&EMAX[dst * 4 + h], lrelu(AS[src * 4 + h] + AD[dst * 4 + h]));
}
__global__ void k_gat_den(const int* __restrict__ eidx, const float* __restrict__ AS,
                          const float* __restrict__ AD, const float* __restrict__ EMAX,
                          float* __restrict__ EDEN, int E, int Nv) {
    int i = blockIdx.x * blockDim.x + threadIdx.x;
    if (i >= E + Nv) return;
    int src, dst; gat_edge(eidx, i, E, src, dst);
    #pragma unroll
    for (int h = 0; h < 4; ++h) {
        float l = lrelu(AS[src * 4 + h] + AD[dst * 4 + h]);
        atomicAdd(&EDEN[dst * 4 + h], __expf(l - EMAX[dst * 4 + h]));
    }
}
__global__ void k_gat_agg(const int* __restrict__ eidx, const float* __restrict__ AS,
                          const float* __restrict__ AD, const float* __restrict__ EMAX,
                          const float* __restrict__ EDEN, const float* __restrict__ XW,
                          float* __restrict__ GOUT, int E, int Nv, int H) {
    int i = blockIdx.x, j = threadIdx.x;
    int src, dst; gat_edge(eidx, i, E, src, dst);
    int h = j >> 6;
    float l = lrelu(AS[src * 4 + h] + AD[dst * 4 + h]);
    float alpha = __expf(l - EMAX[dst * 4 + h]) / EDEN[dst * 4 + h];
    atomicAdd(&GOUT[(size_t)dst * H + j], alpha * XW[(size_t)src * H + j]);
}
// Y = tanh(X + G + b) in place into X
__global__ void k_gat_final(float* __restrict__ X, const float* __restrict__ G,
                            const float* __restrict__ b, int H) {
    size_t base = (size_t)blockIdx.x * H + threadIdx.x;
    X[base] = tanhf(X[base] + G[base] + b[threadIdx.x]);
}

// ---------------- final small heads ----------------
__global__ void k_dot_sig(const float* __restrict__ X, const float* __restrict__ W2,
                          const float* __restrict__ b2, float* __restrict__ out, int H) {
    __shared__ float sm[256];
    float s = block_sum(X[(size_t)blockIdx.x * H + threadIdx.x] * W2[threadIdx.x], sm);
    if (threadIdx.x == 0) out[blockIdx.x] = 1.f / (1.f + __expf(-(s + b2[0])));
}
__global__ void k_combine(const float* Sg, const float* Sl, const float* Dg, const float* Dl,
                          float* Se, float* De, int n) {
    int e = blockIdx.x * blockDim.x + threadIdx.x;
    if (e >= n) return;
    Se[e] = 0.5f * Sg[e] + 0.5f * Sl[e];
    De[e] = 0.5f * Dg[e] + 0.5f * Dl[e];
}

// ---------------- host launcher ----------------
extern "C" void kernel_launch(void* const* d_in, const int* in_sizes, int n_in,
                              void* d_out, int out_size, void* d_ws, size_t ws_size,
                              hipStream_t stream) {
    const float* Xg = (const float*)d_in[0];
    const float* Xl = (const float*)d_in[1];
    const float* Xe = (const float*)d_in[2];
    const int* C_vertex = (const int*)d_in[3];
    const int* C_edge = (const int*)d_in[4];
    const int* T_vertex = (const int*)d_in[5];
    const int* H_vertex = (const int*)d_in[6];
    const int* T_edge = (const int*)d_in[7];
    const int* H_edge = (const int*)d_in[8];
    const int* e_index = (const int*)d_in[9];
    const float* ling_W = (const float*)d_in[10];  const float* ling_b = (const float*)d_in[11];
    const float* linl_W = (const float*)d_in[12];  const float* linl_b = (const float*)d_in[13];
    const float* mha_in_W = (const float*)d_in[14]; const float* mha_in_b = (const float*)d_in[15];
    const float* mha_out_W = (const float*)d_in[16]; const float* mha_out_b = (const float*)d_in[17];
    const float* normg_w = (const float*)d_in[18]; const float* normg_b = (const float*)d_in[19];
    const float* gn_w = (const float*)d_in[20]; const float* gn_b = (const float*)d_in[21];
    const float* gn_ms = (const float*)d_in[22];
    const float* gat_W = (const float*)d_in[23];
    const float* gat_att_src = (const float*)d_in[24];
    const float* gat_att_dst = (const float*)d_in[25];
    const float* gat_b = (const float*)d_in[26];
    const float* lingS_W = (const float*)d_in[27]; const float* lingS_b = (const float*)d_in[28];
    const float* linlS_W = (const float*)d_in[29]; const float* linlS_b = (const float*)d_in[30];
    const float* lingD_W = (const float*)d_in[31]; const float* linlD_W = (const float*)d_in[32];
    const float* mlp_W1 = (const float*)d_in[33]; const float* mlp_b1 = (const float*)d_in[34];
    const float* mlp_ln_w = (const float*)d_in[35]; const float* mlp_ln_b = (const float*)d_in[36];
    const float* mlp_W2 = (const float*)d_in[37]; const float* mlp_b2 = (const float*)d_in[38];
    float* out = (float*)d_out;   // [Pe | Se | De], NEc each

    // workspace layout
    float* wsp = (float*)d_ws;
    size_t off = 0;
    auto alloc = [&](size_t n) { float* p = wsp + off; off += (n + 63) & ~(size_t)63; return p; };
    float* XgA  = alloc((size_t)NGc * Hc);
    float* QKV  = alloc((size_t)NGc * 3 * Hc);
    float* OH   = alloc((size_t)NGc * Hc);
    float* S    = alloc((size_t)NGc * NGc);     // per-head scores; reused as MLP hidden
    float* OP   = alloc((size_t)NGc * Hc);
    float* XgF  = alloc((size_t)NGc * Hc);
    float* MT   = alloc((size_t)NEc * Hc);
    float* MH   = alloc((size_t)NEc * Hc);
    float* ABUF = alloc((size_t)NEc * Hc);
    float* XlA  = alloc((size_t)NVc * Hc);
    float* XLG  = alloc((size_t)NVc * Hc);
    float* XW   = alloc((size_t)NVc * Hc);
    float* GOUT = alloc((size_t)NVc * Hc);
    float* cntT = alloc(NEc);
    float* cntH = alloc(NEc);
    float* accT = alloc(NEc);
    float* accHm = alloc(NEc);
    float* accTH = alloc(NEc);
    float* accHT = alloc(NEc);
    float* SegB = alloc(NEc);
    float* DegB = alloc(NEc);
    float* SelB = alloc(NEc);
    float* DelB = alloc(NEc);
    float* AS   = alloc((size_t)NVc * 4);
    float* AD   = alloc((size_t)NVc * 4);
    float* EMAX = alloc((size_t)NVc * 4);
    float* EDEN = alloc((size_t)NVc * 4);
    float* GMEAN = alloc((size_t)NGRAPHc * Hc);
    float* GVAR  = alloc((size_t)NGRAPHc * Hc);
    float* GCNT  = alloc(NGRAPHc);

    // generic GEMM (B is NxK, runtime strides)
    auto gemm = [&](const float* A, int lda, const float* B, int ldb,
                    const float* bias, float* Y, int ldy, int M, int N, int K,
                    float scale, int act) {
        dim3 g(N / 64, (M + 63) / 64);
        k_gemm_t<0, 0><<<g, 128, 0, stream>>>(A, lda, B, ldb, bias, Y, ldy, M, N, K, scale, act);
    };
    // attention*V GEMM (B is K-major with compile-time stride 3*H)
    auto gemm_av = [&](const float* A, int lda, const float* B,
                       float* Y, int ldy, int M, int N, int K) {
        dim3 g(N / 64, (M + 63) / 64);
        k_gemm_t<1, 3 * Hc><<<g, 128, 0, stream>>>(A, lda, B, 3 * Hc, nullptr, Y, ldy,
                                                   M, N, K, 1.f, 0);
    };
    auto fill = [&](float* p, float v, long n) {
        k_fill<<<(unsigned)((n + 255) / 256), 256, 0, stream>>>(p, v, n);
    };

    // ===== global branch =====
    gemm(Xg, DGc, ling_W, DGc, ling_b, XgA, Hc, NGc, Hc, DGc, 1.f, 1);            // tanh(linear)
    gemm(XgA, Hc, mha_in_W, Hc, mha_in_b, QKV, 3 * Hc, NGc, 3 * Hc, Hc, 1.f, 0);  // qkv
    for (int h = 0; h < HEADSc; ++h) {
        gemm(QKV + h * DHc, 3 * Hc, QKV + Hc + h * DHc, 3 * Hc, nullptr,
             S, NGc, NGc, NGc, DHc, 0.125f, 0);                                   // q·k^T / sqrt(D)
        k_softmax<<<NGc, 256, 0, stream>>>(S, NGc);
        gemm_av(S, NGc, QKV + 2 * Hc + h * DHc, OH + h * DHc, Hc, NGc, DHc, NGc); // attn·v
    }
    gemm(OH, Hc, mha_out_W, Hc, mha_out_b, OP, Hc, NGc, Hc, Hc, 1.f, 0);
    k_add_ln_tanh<<<NGc, Hc, 0, stream>>>(XgA, OP, normg_w, normg_b, XgF, Hc);

    // ----- global scores -----
    fill(MT, 0.f, (long)NEc * Hc); fill(MH, 0.f, (long)NEc * Hc);
    fill(cntT, 0.f, NEc); fill(cntH, 0.f, NEc);
    fill(accT, 0.f, NEc); fill(accHm, 0.f, NEc); fill(accTH, 0.f, NEc); fill(accHT, 0.f, NEc);
    k_edge_mean_acc<<<NTc, Hc, 0, stream>>>(XgF, C_vertex, T_vertex, T_edge, MT, cntT, Hc);
    k_edge_mean_acc<<<NHc, Hc, 0, stream>>>(XgF, C_vertex, H_vertex, H_edge, MH, cntH, Hc);
    k_div_rows<<<NEc, Hc, 0, stream>>>(MT, cntT, Hc);
    k_div_rows<<<NEc, Hc, 0, stream>>>(MH, cntH, Hc);
    k_score_var<<<NTc, Hc, 0, stream>>>(XgF, C_vertex, T_vertex, T_edge, MT, MH,
                                        lingS_W, 0, 2 * Hc, accT, accTH, Hc);
    k_score_var<<<NHc, Hc, 0, stream>>>(XgF, C_vertex, H_vertex, H_edge, MH, MT,
                                        lingS_W, Hc, 3 * Hc, accHm, accHT, Hc);
    k_se_final<<<(NEc + 255) / 256, 256, 0, stream>>>(accT, accHm, accTH, accHT,
                                                      cntT, cntH, lingS_b, SegB, NEc);
    gemm(MT, Hc, lingD_W, Hc, nullptr, ABUF, Hc, NEc, Hc, Hc, 1.f, 0);
    k_cos_de<<<NEc, Hc, 0, stream>>>(ABUF, MH, DegB, Hc);

    // ===== local branch =====
    gemm(Xl, DLc, linl_W, DLc, linl_b, XlA, Hc, NVc, Hc, DLc, 1.f, 1);            // tanh(linear)
    fill(GMEAN, 0.f, (long)NGRAPHc * Hc); fill(GVAR, 0.f, (long)NGRAPHc * Hc);
    fill(GCNT, 0.f, NGRAPHc);
    k_gn_acc<<<NVc, Hc, 0, stream>>>(XlA, C_edge, GMEAN, GCNT, Hc);
    k_div_rows<<<NGRAPHc, Hc, 0, stream>>>(GMEAN, GCNT, Hc);
    k_gn_var<<<NVc, Hc, 0, stream>>>(XlA, C_edge, GMEAN, gn_ms, GVAR, Hc);
    k_div_rows<<<NGRAPHc, Hc, 0, stream>>>(GVAR, GCNT, Hc);
    k_gn_final<<<NVc, Hc, 0, stream>>>(XlA, C_edge, GMEAN, GVAR, gn_ms, gn_w, gn_b, XLG, Hc);

    // GAT
    gemm(XLG, Hc, gat_W, Hc, nullptr, XW, Hc, NVc, Hc, Hc, 1.f, 0);
    k_gat_coef<<<NVc, Hc, 0, stream>>>(XW, gat_att_src, gat_att_dst, AS, AD, Hc);
    fill(EMAX, -3.402823e38f, (long)NVc * 4);
    fill(EDEN, 0.f, (long)NVc * 4);
    fill(GOUT, 0.f, (long)NVc * Hc);
    int E2 = EDGc + NVc;
    k_gat_max<<<(E2 + 255) / 256, 256, 0, stream>>>(e_index, AS, AD, EMAX, EDGc, NVc);
    k_gat_den<<<(E2 + 255) / 256, 256, 0, stream>>>(e_index, AS, AD, EMAX, EDEN, EDGc, NVc);
    k_gat_agg<<<E2, Hc, 0, stream>>>(e_index, AS, AD, EMAX, EDEN, XW, GOUT, EDGc, NVc, Hc);
    k_gat_final<<<NVc, Hc, 0, stream>>>(XLG, GOUT, gat_b, Hc);   // XLG <- tanh(XLG+gat+b)

    // ----- local scores (reuse buffers) -----
    fill(MT, 0.f, (long)NEc * Hc); fill(MH, 0.f, (long)NEc * Hc);
    fill(cntT, 0.f, NEc); fill(cntH, 0.f, NEc);
    fill(accT, 0.f, NEc); fill(accHm, 0.f, NEc); fill(accTH, 0.f, NEc); fill(accHT, 0.f, NEc);
    k_edge_mean_acc<<<NTc, Hc, 0, stream>>>(XLG, nullptr, T_vertex, T_edge, MT, cntT, Hc);
    k_edge_mean_acc<<<NHc, Hc, 0, stream>>>(XLG, nullptr, H_vertex, H_edge, MH, cntH, Hc);
    k_div_rows<<<NEc, Hc, 0, stream>>>(MT, cntT, Hc);
    k_div_rows<<<NEc, Hc, 0, stream>>>(MH, cntH, Hc);
    k_score_var<<<NTc, Hc, 0, stream>>>(XLG, nullptr, T_vertex, T_edge, MT, MH,
                                        linlS_W, 0, 2 * Hc, accT, accTH, Hc);
    k_score_var<<<NHc, Hc, 0, stream>>>(XLG, nullptr, H_vertex, H_edge, MH, MT,
                                        linlS_W, Hc, 3 * Hc, accHm, accHT, Hc);
    k_se_final<<<(NEc + 255) / 256, 256, 0, stream>>>(accT, accHm, accTH, accHT,
                                                      cntT, cntH, linlS_b, SelB, NEc);
    gemm(MT, Hc, linlD_W, Hc, nullptr, ABUF, Hc, NEc, Hc, Hc, 1.f, 0);
    k_cos_de<<<NEc, Hc, 0, stream>>>(ABUF, MH, DelB, Hc);

    // combine Se / De -> out
    k_combine<<<(NEc + 255) / 256, 256, 0, stream>>>(SegB, SelB, DegB, DelB,
                                                     out + NEc, out + 2 * NEc, NEc);

    // ===== MLP head (Pe) ===== (reuse S slab as hidden buffer)
    float* H1 = S;
    gemm(Xe, DGc, mlp_W1, DGc, mlp_b1, H1, HIDc, NEc, HIDc, DGc, 1.f, 2);  // relu
    k_ln_rows<<<NEc, HIDc, 0, stream>>>(H1, mlp_ln_w, mlp_ln_b, HIDc);
    k_dot_sig<<<NEc, HIDc, 0, stream>>>(H1, mlp_W2, mlp_b2, out, HIDc);

    (void)in_sizes; (void)n_in; (void)out_size; (void)ws_size;
}
